// Attention_29755533427504
// MI455X (gfx1250) — compile-verified, matrix operand order checked by reading, not run on hardware
//
#include <hip/hip_runtime.h>
#include <hip/hip_bf16.h>
#include <stdint.h>

// ---------------------------------------------------------------------------
// Problem constants (from reference): b=2, n=2048, dim=1024, h=16, d=64
// ---------------------------------------------------------------------------
#define SEQ   2048
#define DIMM  1024
#define HEADS 16
#define HD    64
#define MTOT  4096      // b * n
#define QKVN  3072      // 3 * dim

typedef _Float16 f16;
typedef __attribute__((ext_vector_type(8)))  _Float16 v8h;
typedef __attribute__((ext_vector_type(16))) _Float16 v16h;
typedef __attribute__((ext_vector_type(8)))  float    v8f;
typedef __attribute__((address_space(3))) _Float16 lds_f16;

struct alignas(8) h4 { f16 x, y, z, w; };

// Pack two contiguous 8-half groups into a 16-half WMMA A/B fragment.
// CDNA5 16-bit A-matrix 16x32 layout: lane(0..15)=row M, lanes 16..31 same M
// with K offset 8; elements 0..7 = K[kb..kb+7], 8..15 = K[16+kb..16+kb+7].
__device__ __forceinline__ v16h pack16(v8h lo, v8h hi) {
    v16h r;
#pragma unroll
    for (int i = 0; i < 8; ++i) { r[i] = lo[i]; r[8 + i] = hi[i]; }
    return r;
}

// p must already point at &src[row*ld + k0 + kb8] (16B aligned)
__device__ __forceinline__ v16h load_ab_frag(const f16* __restrict__ p) {
    return pack16(*(const v8h*)p, *(const v8h*)(p + 16));
}

__device__ __forceinline__ v8f wmma_f16(v16h a, v16h b, v8f c) {
    return __builtin_amdgcn_wmma_f32_16x16x32_f16(
        /*neg_a=*/false, a, /*neg_b=*/false, b,
        /*c_mod=*/(short)0, c, /*reuse_a=*/false, /*reuse_b=*/false);
}

// Async global->LDS copy, 16B per lane, tracked on ASYNCcnt
// (GLOBAL_LOAD_ASYNC_TO_LDS_B128, CDNA5 ISA)
__device__ __forceinline__ void async_ldg_b128(const f16* gsrc, lds_f16* ldst) {
    asm volatile("global_load_async_to_lds_b128 %0, %1, off"
                 :: "v"(ldst), "v"(gsrc) : "memory");
}
__device__ __forceinline__ void wait_async0() {
    asm volatile("s_wait_asynccnt 0x0" ::: "memory");
}

// ---------------------------------------------------------------------------
// Kernel 0: f32 -> f16 conversion, 4 elements / thread
// ---------------------------------------------------------------------------
__global__ void cvt_f32_to_f16_x4(const float* __restrict__ s,
                                  f16* __restrict__ d, int n4) {
    int i = blockIdx.x * blockDim.x + threadIdx.x;
    if (i < n4) {
        float4 v = reinterpret_cast<const float4*>(s)[i];
        h4 r;
        r.x = (f16)v.x; r.y = (f16)v.y; r.z = (f16)v.z; r.w = (f16)v.w;
        reinterpret_cast<h4*>(d)[i] = r;
    }
}

// ---------------------------------------------------------------------------
// Kernel 1: qkv = x @ Wqkv^T  (NT GEMM, f16 in / f32 acc), fused RoPE epilogue.
// All 8 waves in a block share one 64-row weight tile: its 32-wide K-chunk is
// staged into LDS each step with async b128 copies (4 KB = 256 thr x 16 B).
// Each wave computes a 32x64 tile; the 64-wide N tile is one head of q|k|v.
// Writes: Q,K as [bh][n][d] f16 (Q pre-scaled by d^-0.5); V as Vt [bh][d][n].
// ---------------------------------------------------------------------------
__global__ __launch_bounds__(256) void qkv_rope_kernel(
    const f16* __restrict__ xh, const f16* __restrict__ wh,
    const float* __restrict__ sinp, const float* __restrict__ cosp,
    f16* __restrict__ Qb, f16* __restrict__ Kb, f16* __restrict__ Vt)
{
    __shared__ __align__(16) _Float16 Blds[64][40];  // pad 40: conflict-free b128 reads
    const int tid  = threadIdx.x;
    const int wave = tid >> 5;
    const int lane = tid & 31;
    const int r16  = lane & 15;
    const int kb8  = (lane >> 4) << 3;
    const int wid  = blockIdx.x * 8 + wave;
    const int mt   = wid & 127;        // 128 M tiles of 32
    const int nt   = wid >> 7;         // 48  N tiles of 64 (same for all waves in block)
    const int m0   = mt * 32;
    const int n0   = nt * 64;
    const int crow = tid >> 2;         // async-copy row   (0..63)
    const int cseg = (tid & 3) * 8;    // async-copy 8-half segment

    v8f acc[2][4] = {};

    for (int k0 = 0; k0 < DIMM; k0 += 32) {
        // stage B chunk: Wqkv rows [n0, n0+64), K cols [k0, k0+32)
        async_ldg_b128(wh + (size_t)(n0 + crow) * DIMM + k0 + cseg,
                       (lds_f16*)&Blds[crow][cseg]);
        wait_async0();
        __syncthreads();

        if (k0 + 64 < DIMM)
            __builtin_prefetch(xh + (size_t)(m0 + r16) * DIMM + k0 + 64, 0, 3);

        v16h a[2], bm[4];
#pragma unroll
        for (int ms = 0; ms < 2; ++ms)
            a[ms] = load_ab_frag(xh + (size_t)(m0 + ms * 16 + r16) * DIMM + k0 + kb8);
#pragma unroll
        for (int ns = 0; ns < 4; ++ns)
            bm[ns] = pack16(*(const v8h*)&Blds[ns * 16 + r16][kb8],
                            *(const v8h*)&Blds[ns * 16 + r16][16 + kb8]);
#pragma unroll
        for (int ms = 0; ms < 2; ++ms)
#pragma unroll
            for (int ns = 0; ns < 4; ++ns)
                acc[ms][ns] = wmma_f16(a[ms], bm[ns], acc[ms][ns]);

        __syncthreads();   // protect LDS before next chunk's async writes
    }

    const int qkv  = nt >> 4;          // 0=q, 1=k, 2=v
    const int head = nt & 15;

    if (qkv == 2) {
        // V: store transposed -> Vt[bh][d][n]
#pragma unroll
        for (int ms = 0; ms < 2; ++ms)
#pragma unroll
            for (int i = 0; i < 8; ++i) {
                int grow = m0 + ms * 16 + kb8 + i;       // global row in [0,4096)
                int bidx = grow >> 11, nseq = grow & (SEQ - 1);
                size_t dbase = (size_t)(bidx * HEADS + head) * HD;
#pragma unroll
                for (int ns = 0; ns < 4; ++ns) {
                    int d = ns * 16 + r16;
                    Vt[(dbase + d) * SEQ + nseq] = (f16)acc[ms][ns][i];
                }
            }
    } else {
        f16* dst = (qkv == 0) ? Qb : Kb;
        const float qscale = (qkv == 0) ? 0.125f : 1.0f;   // d^-0.5 folded into Q
#pragma unroll
        for (int ms = 0; ms < 2; ++ms)
#pragma unroll
            for (int i = 0; i < 8; ++i) {
                int grow = m0 + ms * 16 + kb8 + i;
                int bidx = grow >> 11, nseq = grow & (SEQ - 1);
                size_t obase = ((size_t)(bidx * HEADS + head) * SEQ + nseq) * HD;
#pragma unroll
                for (int ns = 0; ns < 4; ++ns) {
                    int d = ns * 16 + r16;
                    float cv = cosp[nseq * HD + d];
                    float sv = sinp[nseq * HD + d];
                    // rotate_half pair d^32 lives in fragment ns^2, same lane
                    float rot = (ns < 2) ? -acc[ms][ns + 2][i] : acc[ms][ns - 2][i];
                    float val = (acc[ms][ns][i] * cv + rot * sv) * qscale;
                    dst[obase + d] = (f16)val;
                }
            }
    }
}

// ---------------------------------------------------------------------------
// Kernel 2: flash attention, non-causal. One wave = 32 query rows, keys
// streamed in blocks of 32. Online softmax (f32); P staged via per-wave LDS
// to convert C-layout -> A-layout for the P@V WMMAs.
// ---------------------------------------------------------------------------
__global__ __launch_bounds__(256) void attn_kernel(
    const f16* __restrict__ Qb, const f16* __restrict__ Kb,
    const f16* __restrict__ Vt, f16* __restrict__ Ob)
{
    __shared__ __align__(16) _Float16 Plds[8][32][40];   // pad 40: 16B-aligned rows
    const int wave = threadIdx.x >> 5;
    const int lane = threadIdx.x & 31;
    const int r16  = lane & 15;
    const int kb8  = (lane >> 4) << 3;
    const int bh   = blockIdx.y;                 // 0..31
    const int bidx = bh >> 4, head = bh & 15;
    const int m0   = (blockIdx.x * 8 + wave) * 32;

    const f16* Qh = Qb + (size_t)bh * SEQ * HD;
    const f16* Kh = Kb + (size_t)bh * SEQ * HD;
    const f16* Vh = Vt + (size_t)bh * HD * SEQ;

    // Q fragments: 2 M-subtiles x 2 K-chunks (d = 64 = 2*32)
    v16h qa[2][2];
#pragma unroll
    for (int ms = 0; ms < 2; ++ms)
#pragma unroll
        for (int kc = 0; kc < 2; ++kc)
            qa[ms][kc] = load_ab_frag(Qh + (size_t)(m0 + ms * 16 + r16) * HD + kc * 32 + kb8);

    v8f o[2][4] = {};
    float mrow[2][8], lrow[2][8];
#pragma unroll
    for (int ms = 0; ms < 2; ++ms)
#pragma unroll
        for (int i = 0; i < 8; ++i) { mrow[ms][i] = -1e30f; lrow[ms][i] = 0.0f; }

    for (int n0 = 0; n0 < SEQ; n0 += 32) {
        // ---- S = Q @ K^T  (NT: both contiguous along d) ----
        v16h kf[2][2];
#pragma unroll
        for (int ns = 0; ns < 2; ++ns)
#pragma unroll
            for (int kc = 0; kc < 2; ++kc)
                kf[ns][kc] = load_ab_frag(Kh + (size_t)(n0 + ns * 16 + r16) * HD + kc * 32 + kb8);
        v8f s[2][2] = {};
#pragma unroll
        for (int ms = 0; ms < 2; ++ms)
#pragma unroll
            for (int ns = 0; ns < 2; ++ns) {
                s[ms][ns] = wmma_f16(qa[ms][0], kf[ns][0], s[ms][ns]);
                s[ms][ns] = wmma_f16(qa[ms][1], kf[ns][1], s[ms][ns]);
            }

        // ---- online softmax (row = lane-half's 8 rows; reduce over 16 lanes) ----
#pragma unroll
        for (int ms = 0; ms < 2; ++ms) {
            float rmax[8];
#pragma unroll
            for (int i = 0; i < 8; ++i) rmax[i] = fmaxf(s[ms][0][i], s[ms][1][i]);
#pragma unroll
            for (int msk = 1; msk < 16; msk <<= 1)
#pragma unroll
                for (int i = 0; i < 8; ++i)
                    rmax[i] = fmaxf(rmax[i], __shfl_xor(rmax[i], msk, 32));
            float fac[8];
#pragma unroll
            for (int i = 0; i < 8; ++i) {
                float nm = fmaxf(mrow[ms][i], rmax[i]);
                fac[i] = __expf(mrow[ms][i] - nm);
                mrow[ms][i] = nm;
            }
            float rsum[8];
#pragma unroll
            for (int i = 0; i < 8; ++i) rsum[i] = 0.0f;
#pragma unroll
            for (int ns = 0; ns < 2; ++ns)
#pragma unroll
                for (int i = 0; i < 8; ++i) {
                    float p = __expf(s[ms][ns][i] - mrow[ms][i]);
                    s[ms][ns][i] = p;
                    rsum[i] += p;
                }
#pragma unroll
            for (int msk = 1; msk < 16; msk <<= 1)
#pragma unroll
                for (int i = 0; i < 8; ++i)
                    rsum[i] += __shfl_xor(rsum[i], msk, 32);
#pragma unroll
            for (int i = 0; i < 8; ++i)
                lrow[ms][i] = lrow[ms][i] * fac[i] + rsum[i];
#pragma unroll
            for (int ds = 0; ds < 4; ++ds)
#pragma unroll
                for (int i = 0; i < 8; ++i) o[ms][ds][i] *= fac[i];
            // stage P (C-layout) into LDS
#pragma unroll
            for (int ns = 0; ns < 2; ++ns)
#pragma unroll
                for (int i = 0; i < 8; ++i)
                    Plds[wave][ms * 16 + kb8 + i][ns * 16 + r16] = (f16)s[ms][ns][i];
        }
        // same-wave LDS RAW: DS ops are in-order per wave; keep compiler honest
        asm volatile("s_wait_dscnt 0" ::: "memory");

        // reload P in A-layout
        v16h pa[2];
#pragma unroll
        for (int ms = 0; ms < 2; ++ms)
            pa[ms] = pack16(*(const v8h*)&Plds[wave][ms * 16 + r16][kb8],
                            *(const v8h*)&Plds[wave][ms * 16 + r16][16 + kb8]);
        // V fragments from Vt[bh][d][n] (contiguous along keys)
        v16h vf[4];
#pragma unroll
        for (int ds = 0; ds < 4; ++ds)
            vf[ds] = load_ab_frag(Vh + (size_t)(ds * 16 + r16) * SEQ + n0 + kb8);
#pragma unroll
        for (int ms = 0; ms < 2; ++ms)
#pragma unroll
            for (int ds = 0; ds < 4; ++ds)
                o[ms][ds] = wmma_f16(pa[ms], vf[ds], o[ms][ds]);
    }

    // ---- epilogue: O /= l, write [b][n][h*64+d] f16 ----
#pragma unroll
    for (int ms = 0; ms < 2; ++ms)
#pragma unroll
        for (int i = 0; i < 8; ++i) {
            float inv = 1.0f / lrow[ms][i];
            int nseq = m0 + ms * 16 + kb8 + i;
            size_t obase = ((size_t)bidx * SEQ + nseq) * DIMM + head * HD;
#pragma unroll
            for (int ds = 0; ds < 4; ++ds)
                Ob[obase + ds * 16 + r16] = (f16)(o[ms][ds][i] * inv);
        }
}

// ---------------------------------------------------------------------------
// Kernel 3: out = O @ Wproj^T + bproj  (NT GEMM, f32 output), async-LDS
// staged weight tile exactly like kernel 1.
// ---------------------------------------------------------------------------
__global__ __launch_bounds__(256) void proj_kernel(
    const f16* __restrict__ Oh, const f16* __restrict__ wh,
    const float* __restrict__ bias, float* __restrict__ out)
{
    __shared__ __align__(16) _Float16 Blds[64][40];
    const int tid  = threadIdx.x;
    const int wave = tid >> 5;
    const int lane = tid & 31;
    const int r16  = lane & 15;
    const int kb8  = (lane >> 4) << 3;
    const int wid  = blockIdx.x * 8 + wave;
    const int mt   = wid & 127;        // 128 M tiles
    const int nt   = wid >> 7;         // 16  N tiles (same for all waves in block)
    const int m0   = mt * 32;
    const int n0   = nt * 64;
    const int crow = tid >> 2;
    const int cseg = (tid & 3) * 8;

    v8f acc[2][4] = {};

    for (int k0 = 0; k0 < DIMM; k0 += 32) {
        async_ldg_b128(wh + (size_t)(n0 + crow) * DIMM + k0 + cseg,
                       (lds_f16*)&Blds[crow][cseg]);
        wait_async0();
        __syncthreads();

        if (k0 + 64 < DIMM)
            __builtin_prefetch(Oh + (size_t)(m0 + r16) * DIMM + k0 + 64, 0, 3);

        v16h a[2], bm[4];
#pragma unroll
        for (int ms = 0; ms < 2; ++ms)
            a[ms] = load_ab_frag(Oh + (size_t)(m0 + ms * 16 + r16) * DIMM + k0 + kb8);
#pragma unroll
        for (int ns = 0; ns < 4; ++ns)
            bm[ns] = pack16(*(const v8h*)&Blds[ns * 16 + r16][kb8],
                            *(const v8h*)&Blds[ns * 16 + r16][16 + kb8]);
#pragma unroll
        for (int ms = 0; ms < 2; ++ms)
#pragma unroll
            for (int ns = 0; ns < 4; ++ns)
                acc[ms][ns] = wmma_f16(a[ms], bm[ns], acc[ms][ns]);

        __syncthreads();
    }

#pragma unroll
    for (int ms = 0; ms < 2; ++ms)
#pragma unroll
        for (int i = 0; i < 8; ++i) {
            int grow = m0 + ms * 16 + kb8 + i;
#pragma unroll
            for (int ns = 0; ns < 4; ++ns) {
                int col = n0 + ns * 16 + r16;
                out[(size_t)grow * DIMM + col] = acc[ms][ns][i] + bias[col];
            }
        }
}

// ---------------------------------------------------------------------------
// Host launcher
// ---------------------------------------------------------------------------
extern "C" void kernel_launch(void* const* d_in, const int* in_sizes, int n_in,
                              void* d_out, int out_size, void* d_ws, size_t ws_size,
                              hipStream_t stream) {
    const float* x     = (const float*)d_in[0];   // [2,2048,1024]
    const float* Wqkv  = (const float*)d_in[1];   // [3072,1024]
    const float* Wproj = (const float*)d_in[2];   // [1024,1024]
    const float* bproj = (const float*)d_in[3];   // [1024]
    const float* sinp  = (const float*)d_in[4];   // [2048,64]
    const float* cosp  = (const float*)d_in[5];   // [2048,64]
    float* out = (float*)d_out;

    char* ws = (char*)d_ws;
    const size_t MB = 1u << 20;
    f16* xh     = (f16*)(ws + 0 * MB);   // 8 MB  (4096x1024)
    f16* wqkvh  = (f16*)(ws + 8 * MB);   // 6 MB  (3072x1024)
    f16* wprojh = (f16*)(ws + 14 * MB);  // 2 MB  (1024x1024)
    f16* Qb     = (f16*)(ws + 16 * MB);  // 8 MB  [bh][n][d]
    f16* Kb     = (f16*)(ws + 24 * MB);  // 8 MB  [bh][n][d]
    f16* Vt     = (f16*)(ws + 32 * MB);  // 8 MB  [bh][d][n]
    f16* Ob     = (f16*)(ws + 40 * MB);  // 8 MB  [b][n][dim]

    // f32 -> f16 converts
    cvt_f32_to_f16_x4<<<(MTOT * DIMM / 4 + 255) / 256, 256, 0, stream>>>(x, xh, MTOT * DIMM / 4);
    cvt_f32_to_f16_x4<<<(QKVN * DIMM / 4 + 255) / 256, 256, 0, stream>>>(Wqkv, wqkvh, QKVN * DIMM / 4);
    cvt_f32_to_f16_x4<<<(DIMM * DIMM / 4 + 255) / 256, 256, 0, stream>>>(Wproj, wprojh, DIMM * DIMM / 4);

    // QKV GEMM + RoPE: 128 M-tiles x 48 N-tiles = 6144 waves / 8 per block
    qkv_rope_kernel<<<768, 256, 0, stream>>>(xh, wqkvh, sinp, cosp, Qb, Kb, Vt);

    // Flash attention: 8 row-blocks x 32 (b*h)
    attn_kernel<<<dim3(8, 32), 256, 0, stream>>>(Qb, Kb, Vt, Ob);

    // Output projection: 128 M-tiles x 16 N-tiles = 2048 waves / 8 per block
    proj_kernel<<<256, 256, 0, stream>>>(Ob, wprojh, bproj, out);
}
